// ResponsePointTransformerProxyModel_51737176047791
// MI455X (gfx1250) — matrix operand order repaired and checked by reading, not compiled, not run on hardware
//
#include <hip/hip_runtime.h>

// ---------------------------------------------------------------------------
// Types for WMMA fragments (gfx1250, wave32)
// ---------------------------------------------------------------------------
typedef __attribute__((ext_vector_type(16))) __bf16 v16bf;
typedef __attribute__((ext_vector_type(8)))  float  v8f;
typedef __attribute__((ext_vector_type(4)))  unsigned int u32x4;

union BF16Frag { v16bf v; u32x4 u[2]; };

// A fragment: 16x32 bf16, A rows striped within a lane.
// ISA layout: lanes 0-15 -> row=lane, K chunks {0..7, 16..23};
//             lanes 16-31 -> row=lane-16, K chunks {8..15, 24..31}.
__device__ __forceinline__ v16bf load_a_frag(const __bf16* sA, int strideE,
                                             int lane, int kbase) {
  int row = lane & 15;
  int k0  = kbase + ((lane < 16) ? 0 : 8);
  const __bf16* p = sA + row * strideE + k0;
  BF16Frag f;
  f.u[0] = *(const u32x4*)(p);        // K = k0 .. k0+7   (16B aligned)
  f.u[1] = *(const u32x4*)(p + 16);   // K = k0+16 .. k0+23
  return f.v;
}

// B fragment: 32x16 bf16 from transposed weights Wt[N][Kin].
// ISA layout: lanes 0-15 hold K {0..7, 16..23} of col=lane; lanes 16-31 hold K {8..15, 24..31}.
__device__ __forceinline__ v16bf load_b_frag(const __bf16* Wt, int Kin,
                                             int colBase, int lane, int kbase) {
  int col = colBase + (lane & 15);
  int k0  = kbase + ((lane < 16) ? 0 : 8);
  const __bf16* p = Wt + (size_t)col * Kin + k0;
  BF16Frag f;
  f.u[0] = *(const u32x4*)(p);
  f.u[1] = *(const u32x4*)(p + 16);
  return f.v;
}

// Two 16x16 output tiles sharing one A fragment per K-step:
// per 2 WMMAs we issue 2 ds_load_b128 (A) + 4 global_load_b128 (B, L2-hot).
__device__ __forceinline__ void gemm_tile2(const __bf16* sA, int strideE,
                                           const __bf16* Wt, int Kin,
                                           int col0, int col1, int lane,
                                           v8f& acc0, v8f& acc1) {
  for (int kb = 0; kb < Kin; kb += 32) {
    v16bf a  = load_a_frag(sA, strideE, lane, kb);
    v16bf b0 = load_b_frag(Wt, Kin, col0, lane, kb);
    v16bf b1 = load_b_frag(Wt, Kin, col1, lane, kb);
    acc0 = __builtin_amdgcn_wmma_f32_16x16x32_bf16(
        false, a, false, b0, (short)0, acc0, false, false);
    acc1 = __builtin_amdgcn_wmma_f32_16x16x32_bf16(
        false, a, false, b1, (short)0, acc1, false, false);
  }
}

// ---------------------------------------------------------------------------
// Weight convert + transpose:  W[Kin][Nout] f32  ->  Wt[Nout][Kin] bf16
// ---------------------------------------------------------------------------
__global__ void convert_wt_kernel(const float* __restrict__ W,
                                  __bf16* __restrict__ Wt,
                                  int Kin, int Nout) {
  int i = blockIdx.x * 256 + threadIdx.x;
  if (i < Kin * Nout) {
    int n = i / Kin, k = i % Kin;
    Wt[i] = (__bf16)W[(size_t)k * Nout + n];
  }
}

// ---------------------------------------------------------------------------
// Brute-force KNN (k=16, includes self).  One thread per query point,
// candidates tiled through LDS, top-16 kept in registers (unrolled insertion).
// ---------------------------------------------------------------------------
__global__ void knn_kernel(const float* __restrict__ xyz,
                           int* __restrict__ knn, int N) {
  __shared__ float s_c[256][3];
  int tiles = N >> 8;
  int b  = blockIdx.x / tiles;
  int tb = blockIdx.x % tiles;
  int qi = tb * 256 + threadIdx.x;

  const float* pq = xyz + ((size_t)b * N + qi) * 3;
  float qx = pq[0], qy = pq[1], qz = pq[2];

  float bd[16]; int bi[16];
#pragma unroll
  for (int t = 0; t < 16; ++t) { bd[t] = 3.0e38f; bi[t] = 0; }

  for (int base = 0; base < N; base += 256) {
    const float* pc = xyz + ((size_t)b * N + base + threadIdx.x) * 3;
    s_c[threadIdx.x][0] = pc[0];
    s_c[threadIdx.x][1] = pc[1];
    s_c[threadIdx.x][2] = pc[2];
    __syncthreads();
    for (int c = 0; c < 256; ++c) {
      float dx = qx - s_c[c][0];
      float dy = qy - s_c[c][1];
      float dz = qz - s_c[c][2];
      float d = dx * dx + dy * dy + dz * dz;
      if (d < bd[15]) {
        float dcur = d; int icur = base + c;
#pragma unroll
        for (int t = 0; t < 16; ++t) {
          if (dcur < bd[t]) {
            float td = bd[t]; int ti = bi[t];
            bd[t] = dcur; bi[t] = icur;
            dcur = td; icur = ti;
          }
        }
      }
    }
    __syncthreads();
  }
#pragma unroll
  for (int t = 0; t < 16; ++t)
    knn[((size_t)b * N + qi) * 16 + t] = bi[t];
}

// ---------------------------------------------------------------------------
// Generic WMMA GEMM:  C[M x N] = A[M x Kin](f32) @ Wt(bf16,[N][Kin])^T
//                               (+ bias) (+ residual)
// Block: 128 threads (4 waves), each wave computes TWO 16x16 tiles sharing A.
// blockIdx.x -> 16-row tile, blockIdx.y -> 128-column group.
// ---------------------------------------------------------------------------
__global__ void gemm_bf16_kernel(const float* __restrict__ A,
                                 const __bf16* __restrict__ Wt,
                                 const float* __restrict__ bias,
                                 const float* __restrict__ residual,
                                 float* __restrict__ C,
                                 int M, int Kin, int N) {
  __shared__ __bf16 sA[16 * (512 + 8)];     // max Kin = 512
  const int strideE = Kin + 8;              // byte stride multiple of 16
  int tid = threadIdx.x;
  int rowBase = blockIdx.x * 16;

  int total = 16 * Kin;
  for (int i = tid; i < total; i += 128) {
    int r = i / Kin, c = i - r * Kin;
    sA[r * strideE + c] = (__bf16)A[(size_t)(rowBase + r) * Kin + c];
  }
  __syncthreads();

  int wave = tid >> 5, lane = tid & 31;
  int n0 = blockIdx.y * 128 + wave * 32;    // wave owns cols [n0, n0+32)
  if (n0 < N) {
    v8f a0 = {}, a1 = {};
    gemm_tile2(sA, strideE, Wt, Kin, n0, n0 + 16, lane, a0, a1);
    int cA = n0 + (lane & 15), cB = cA + 16;
    float bvA = bias ? bias[cA] : 0.0f;
    float bvB = bias ? bias[cB] : 0.0f;
    int rofs = (lane < 16) ? 0 : 8;
#pragma unroll
    for (int r = 0; r < 8; ++r) {
      int row = rowBase + r + rofs;
      float vA = a0[r] + bvA;
      float vB = a1[r] + bvB;
      if (residual) {
        vA += residual[(size_t)row * N + cA];
        vB += residual[(size_t)row * N + cB];
      }
      C[(size_t)row * N + cA] = vA;
      C[(size_t)row * N + cB] = vB;
    }
  }
}

// ---------------------------------------------------------------------------
// Fused per-point attention: one block (256 thr / 8 waves) per (b,n).
//   t1   = relu(rel @ Wd1 + bd1)              [16,256]  (VALU, K=3)
//   pos  = t1 @ Wd2 + bd2                     [16,256]  (WMMA)
//   h    = q - k_gather + pos                 [16,256]
//   g1   = relu(h @ Wg1 + bg1)                [16,256]  (WMMA)
//   lg   = g1 @ Wg2 + bg2                     [16,256]  (WMMA)
//   attn = softmax over the 16 neighbors (axis=-2), scaled by 1/sqrt(256)
//   res  = sum_k attn * (v_gather + pos)      [256]
// ---------------------------------------------------------------------------
#define DM 256
__global__ void fused_attn_kernel(const float* __restrict__ xyz,
                                  const int* __restrict__ knn,
                                  const float* __restrict__ q,
                                  const float* __restrict__ xk,
                                  const float* __restrict__ xv,
                                  const float* __restrict__ Wd1,
                                  const float* __restrict__ bd1,
                                  const __bf16* __restrict__ Wd2t,
                                  const float* __restrict__ bd2,
                                  const __bf16* __restrict__ Wg1t,
                                  const float* __restrict__ bg1,
                                  const __bf16* __restrict__ Wg2t,
                                  const float* __restrict__ bg2,
                                  float* __restrict__ attn_out,
                                  float* __restrict__ res_mid,
                                  int N) {
  __shared__ float  s_pos[16][DM + 1];
  __shared__ float  s_vpe[16][DM + 1];
  __shared__ float  s_att[16][DM + 1];
  __shared__ __bf16 s_a[16][DM + 8];
  __shared__ float  s_q[DM];
  __shared__ float  s_rel[16][3];
  __shared__ int    s_idx[16];

  int pid = blockIdx.x;                  // b*N + n
  int b = pid / N;
  int tid = threadIdx.x;
  int wave = tid >> 5, lane = tid & 31;
  int n0 = wave * 32;                    // this wave's two column tiles
  int cA = n0 + (lane & 15), cB = cA + 16;
  int rofs = (lane < 16) ? 0 : 8;

  if (tid < 16) {
    int j = knn[(size_t)pid * 16 + tid];
    s_idx[tid] = j;
    const float* pq = xyz + (size_t)pid * 3;
    const float* pj = xyz + ((size_t)b * N + j) * 3;
    s_rel[tid][0] = pq[0] - pj[0];
    s_rel[tid][1] = pq[1] - pj[1];
    s_rel[tid][2] = pq[2] - pj[2];
  }
  s_q[tid] = q[(size_t)pid * DM + tid];
  __syncthreads();

  // t1 = relu(rel @ Wd1 + bd1) -> s_a (bf16)
  {
    int j  = tid >> 4;
    int c0 = tid & 15;
    float r0 = s_rel[j][0], r1 = s_rel[j][1], r2 = s_rel[j][2];
#pragma unroll
    for (int i = 0; i < 16; ++i) {
      int c = c0 + i * 16;
      float v = bd1[c] + r0 * Wd1[c] + r1 * Wd1[DM + c] + r2 * Wd1[2 * DM + c];
      s_a[j][c] = (__bf16)fmaxf(v, 0.0f);
    }
  }
  __syncthreads();

  // GEMM1: pos = s_a @ Wd2 + bd2  (each wave owns 2 of the 16 col-tiles)
  {
    v8f a0 = {}, a1 = {};
    gemm_tile2(&s_a[0][0], DM + 8, Wd2t, DM, n0, n0 + 16, lane, a0, a1);
#pragma unroll
    for (int r = 0; r < 8; ++r) {
      s_pos[r + rofs][cA] = a0[r] + bd2[cA];
      s_pos[r + rofs][cB] = a1[r] + bd2[cB];
    }
  }
  __syncthreads();

  // Gather k/v rows; h = q - k + pos -> s_a (bf16); vpe = v + pos -> s_vpe
  for (int j = 0; j < 16; ++j) {
    size_t ro = ((size_t)b * N + s_idx[j]) * DM + tid;
    float p = s_pos[j][tid];
    s_a[j][tid]   = (__bf16)(s_q[tid] - xk[ro] + p);
    s_vpe[j][tid] = xv[ro] + p;
  }
  __syncthreads();

  // GEMM2: g1 = relu(s_a @ Wg1 + bg1) -> s_a (in place: barrier between RW)
  {
    v8f a0 = {}, a1 = {};
    gemm_tile2(&s_a[0][0], DM + 8, Wg1t, DM, n0, n0 + 16, lane, a0, a1);
    __syncthreads();   // all K-loop reads of s_a complete before overwrite
#pragma unroll
    for (int r = 0; r < 8; ++r) {
      s_a[r + rofs][cA] = (__bf16)fmaxf(a0[r] + bg1[cA], 0.0f);
      s_a[r + rofs][cB] = (__bf16)fmaxf(a1[r] + bg1[cB], 0.0f);
    }
  }
  __syncthreads();

  // GEMM3: logits = s_a @ Wg2 + bg2 -> s_att (f32)
  {
    v8f a0 = {}, a1 = {};
    gemm_tile2(&s_a[0][0], DM + 8, Wg2t, DM, n0, n0 + 16, lane, a0, a1);
#pragma unroll
    for (int r = 0; r < 8; ++r) {
      s_att[r + rofs][cA] = a0[r] + bg2[cA];
      s_att[r + rofs][cB] = a1[r] + bg2[cB];
    }
  }
  __syncthreads();

  // Softmax over the 16 neighbors per channel (thread = channel), then reduce.
  {
    const float scale = 0.0625f;  // 1/sqrt(256)
    float m = -3.0e38f;
#pragma unroll
    for (int j = 0; j < 16; ++j) m = fmaxf(m, s_att[j][tid]);
    float e[16];
    float sum = 0.0f;
#pragma unroll
    for (int j = 0; j < 16; ++j) {
      e[j] = __expf((s_att[j][tid] - m) * scale);
      sum += e[j];
    }
    float inv = 1.0f / sum;
    float racc = 0.0f;
#pragma unroll
    for (int j = 0; j < 16; ++j) {
      float p = e[j] * inv;
      attn_out[((size_t)pid * 16 + j) * DM + tid] = p;
      racc += p * s_vpe[j][tid];
    }
    res_mid[(size_t)pid * DM + tid] = racc;
  }
}

// ---------------------------------------------------------------------------
// Host orchestration
// ---------------------------------------------------------------------------
extern "C" void kernel_launch(void* const* d_in, const int* in_sizes, int n_in,
                              void* d_out, int out_size, void* d_ws, size_t ws_size,
                              hipStream_t stream) {
  const int B = 4, N = 4096, DP = 512, D = 256;
  const int M = B * N;

  const float* xyz  = (const float*)d_in[0];
  const float* feat = (const float*)d_in[1];
  const float* W1   = (const float*)d_in[2];
  const float* b1   = (const float*)d_in[3];
  const float* W2   = (const float*)d_in[4];
  const float* b2   = (const float*)d_in[5];
  const float* Wd1  = (const float*)d_in[6];
  const float* bd1  = (const float*)d_in[7];
  const float* Wd2  = (const float*)d_in[8];
  const float* bd2  = (const float*)d_in[9];
  const float* Wg1  = (const float*)d_in[10];
  const float* bg1  = (const float*)d_in[11];
  const float* Wg2  = (const float*)d_in[12];
  const float* bg2  = (const float*)d_in[13];
  const float* Wq   = (const float*)d_in[14];
  const float* Wk   = (const float*)d_in[15];
  const float* Wv   = (const float*)d_in[16];
  (void)in_sizes; (void)n_in; (void)out_size; (void)ws_size;

  char* ws = (char*)d_ws;
  size_t off = 0;
  auto take = [&](size_t bytes) -> char* {
    char* p = ws + off;
    off += (bytes + 255) & ~(size_t)255;
    return p;
  };

  int*    knn  = (int*)   take((size_t)M * 16 * sizeof(int));
  float*  x    = (float*) take((size_t)M * D * sizeof(float));
  float*  qb   = (float*) take((size_t)M * D * sizeof(float));
  float*  kbuf = (float*) take((size_t)M * D * sizeof(float));
  float*  vbuf = (float*) take((size_t)M * D * sizeof(float));
  float*  rmid = (float*) take((size_t)M * D * sizeof(float));
  __bf16* W1t  = (__bf16*)(void*)take((size_t)D  * DP * 2);
  __bf16* Wqt  = (__bf16*)(void*)take((size_t)D  * D  * 2);
  __bf16* Wkt  = (__bf16*)(void*)take((size_t)D  * D  * 2);
  __bf16* Wvt  = (__bf16*)(void*)take((size_t)D  * D  * 2);
  __bf16* Wd2t = (__bf16*)(void*)take((size_t)D  * D  * 2);
  __bf16* Wg1t = (__bf16*)(void*)take((size_t)D  * D  * 2);
  __bf16* Wg2t = (__bf16*)(void*)take((size_t)D  * D  * 2);
  __bf16* W2t  = (__bf16*)(void*)take((size_t)DP * D  * 2);

  float* out_res  = (float*)d_out;                       // [B,N,512]
  float* out_attn = (float*)d_out + (size_t)M * DP;      // [B,N,16,256]

  // 1) weight conversion (bf16, transposed)
  convert_wt_kernel<<<(DP * D + 255) / 256, 256, 0, stream>>>(W1,  W1t,  DP, D);
  convert_wt_kernel<<<(D * D + 255) / 256, 256, 0, stream>>>(Wq,  Wqt,  D, D);
  convert_wt_kernel<<<(D * D + 255) / 256, 256, 0, stream>>>(Wk,  Wkt,  D, D);
  convert_wt_kernel<<<(D * D + 255) / 256, 256, 0, stream>>>(Wv,  Wvt,  D, D);
  convert_wt_kernel<<<(D * D + 255) / 256, 256, 0, stream>>>(Wd2, Wd2t, D, D);
  convert_wt_kernel<<<(D * D + 255) / 256, 256, 0, stream>>>(Wg1, Wg1t, D, D);
  convert_wt_kernel<<<(D * D + 255) / 256, 256, 0, stream>>>(Wg2, Wg2t, D, D);
  convert_wt_kernel<<<(D * DP + 255) / 256, 256, 0, stream>>>(W2,  W2t,  D, DP);

  // 2) knn
  knn_kernel<<<B * (N / 256), 256, 0, stream>>>(xyz, knn, N);

  // 3) x = feat @ W1 + b1
  gemm_bf16_kernel<<<dim3(M / 16, D / 128), 128, 0, stream>>>(
      feat, W1t, b1, nullptr, x, M, DP, D);

  // 4) q/k/v projections (no bias)
  gemm_bf16_kernel<<<dim3(M / 16, D / 128), 128, 0, stream>>>(
      x, Wqt, nullptr, nullptr, qb, M, D, D);
  gemm_bf16_kernel<<<dim3(M / 16, D / 128), 128, 0, stream>>>(
      x, Wkt, nullptr, nullptr, kbuf, M, D, D);
  gemm_bf16_kernel<<<dim3(M / 16, D / 128), 128, 0, stream>>>(
      x, Wvt, nullptr, nullptr, vbuf, M, D, D);

  // 5) fused per-point attention (writes attn output + res_mid)
  fused_attn_kernel<<<M, 256, 0, stream>>>(
      xyz, knn, qb, kbuf, vbuf, Wd1, bd1, Wd2t, bd2,
      Wg1t, bg1, Wg2t, bg2, out_attn, rmid, N);

  // 6) res = rmid @ W2 + b2 + feat
  gemm_bf16_kernel<<<dim3(M / 16, DP / 128), 128, 0, stream>>>(
      rmid, W2t, b2, feat, out_res, M, D, DP);
}
